// SpikingNeuralNetwork_84885733638297
// MI455X (gfx1250) — matrix operand order repaired and checked by reading, not compiled, not run on hardware
//
#include <hip/hip_runtime.h>

// Problem dims (fixed by the reference)
#define B_ 32
#define S_ 1024
#define I_ 512
#define H_ 2048
#define O_ 512

#define DECAY_ 0.95f
#define THRESH_ 1.0f
#define REFRAC_ 2.0f
#define LATSCL_ 0.1f

#define NB_H   (H_ / 16)          // 128 hidden-column blocks
#define NB_O   (O_ / 16)          // 32 output-column blocks
#define NBLK   (NB_H + NB_O)      // 160 persistent blocks, 1 wave each

typedef __attribute__((ext_vector_type(16))) __bf16 v16bf;
typedef __attribute__((ext_vector_type(8)))  __bf16 v8bf;
typedef __attribute__((ext_vector_type(8)))  float  v8f;

// ---------------------------------------------------------------------------
// Fragment loaders (ISA 7.12.2 layouts).
// A (16x32 bf16): lanes 0-15 -> M=lane,  K {kk+0..7, kk+16..23}
//                 lanes 16-31 -> M=lane-16, K {kk+8..15, kk+24..31}
// B (32x16 bf16): lane&15 = N column, K same as A. Weights pre-transposed
// to [N][K] so a B fragment is two contiguous 16B reads per lane.
// ---------------------------------------------------------------------------
__device__ __forceinline__ v16bf load_a_from_f32(const float* __restrict__ row,
                                                 int kk, int lane) {
  int half = (lane >> 4) & 1;
  int kb = kk + half * 8;
  v16bf a;
#pragma unroll
  for (int j = 0; j < 8; ++j) a[j]     = (__bf16)row[kb + j];
#pragma unroll
  for (int j = 0; j < 8; ++j) a[8 + j] = (__bf16)row[kb + 16 + j];
  return a;
}

__device__ __forceinline__ v16bf load_a_bf16(const __bf16* __restrict__ row,
                                             int kk, int lane) {
  int half = (lane >> 4) & 1;
  const __bf16* p = row + kk + half * 8;
  v8bf lo = *(const v8bf*)(p);        // K = kb .. kb+7
  v8bf hi = *(const v8bf*)(p + 16);   // K = kb+16 .. kb+23
  return __builtin_shufflevector(lo, hi, 0, 1, 2, 3, 4, 5, 6, 7,
                                 8, 9, 10, 11, 12, 13, 14, 15);
}

__device__ __forceinline__ v16bf load_b_t(const __bf16* __restrict__ WT, int ldk,
                                          int n0, int kk, int lane) {
  int col = n0 + (lane & 15);
  int kh  = (lane >> 4) & 1;
  const __bf16* p = WT + (size_t)col * ldk + kk + kh * 8;
  v8bf lo = *(const v8bf*)(p);
  v8bf hi = *(const v8bf*)(p + 16);
  return __builtin_shufflevector(lo, hi, 0, 1, 2, 3, 4, 5, 6, 7,
                                 8, 9, 10, 11, 12, 13, 14, 15);
}

// ---------------------------------------------------------------------------
// Software grid barrier (generation counting, agent scope). All 160 blocks are
// single-wave with 16KB LDS -> trivially co-resident on MI455X, so spinning is
// safe. Acquire polling invalidates stale WGP$ lines; the ACQ_REL arrival
// releases this block's spike writes.
// ---------------------------------------------------------------------------
__device__ __forceinline__ void grid_barrier(unsigned* cnt, unsigned* gen,
                                             unsigned nblocks, unsigned target) {
  __syncthreads();
  if (threadIdx.x == 0) {
    unsigned prev = __hip_atomic_fetch_add(cnt, 1u, __ATOMIC_ACQ_REL,
                                           __HIP_MEMORY_SCOPE_AGENT);
    if (prev == nblocks - 1u) {
      __hip_atomic_store(cnt, 0u, __ATOMIC_RELAXED, __HIP_MEMORY_SCOPE_AGENT);
      __hip_atomic_fetch_add(gen, 1u, __ATOMIC_RELEASE, __HIP_MEMORY_SCOPE_AGENT);
    }
  }
  while (__hip_atomic_load(gen, __ATOMIC_ACQUIRE, __HIP_MEMORY_SCOPE_AGENT) <
         target) {
    __builtin_amdgcn_s_sleep(2);
  }
  __syncthreads();
}

// ---------------------------------------------------------------------------
// Prep kernels
// ---------------------------------------------------------------------------
__global__ void convert_transpose_kernel(const float* __restrict__ W,
                                         __bf16* __restrict__ WT,
                                         int K, int N) {
  long long idx = (long long)blockIdx.x * blockDim.x + threadIdx.x;
  if (idx < (long long)K * N) {
    int k = (int)(idx / N);
    int n = (int)(idx % N);
    WT[(size_t)n * K + k] = (__bf16)W[idx];
  }
}

__global__ void zero_u32_kernel(unsigned* __restrict__ p, int n) {
  int i = blockIdx.x * blockDim.x + threadIdx.x;
  if (i < n) p[i] = 0u;
}

// ---------------------------------------------------------------------------
// Persistent SNN scan kernel. Blocks 0..127 own 16 hidden columns each
// (phase A: ic GEMM + LIF update, phase B: lateral GEMM -> mem update).
// Blocks 128..159 own 16 output columns each (phase B: spikes @ W_h -> out).
// mem/refr live in registers for the whole 1024-step scan.
// ---------------------------------------------------------------------------
__global__ __launch_bounds__(32)
void snn_persistent(const float* __restrict__ x,        // [B,S,I]
                    const __bf16* __restrict__ WinT,    // [H][I]
                    const __bf16* __restrict__ LT,      // [H][H]
                    const __bf16* __restrict__ WhT,     // [O][H]
                    __bf16* __restrict__ spikes,        // [B][H]
                    float* __restrict__ out,            // [B,S,O]
                    unsigned* __restrict__ cnt,
                    unsigned* __restrict__ gen) {
  const int lane = threadIdx.x;
  const int bid  = blockIdx.x;
  const bool is_h = bid < NB_H;                     // uniform per block
  const int n0 = is_h ? bid * 16 : (bid - NB_H) * 16;
  const int n  = n0 + (lane & 15);
  const int mbase = (lane >> 4) * 8;

  // Time-invariant W_in tile for this block's 16 columns: park it in LDS.
  // Layout: [iter 0..15][lane 0..31] x 16 bf16 = 16 KB.
  __shared__ __bf16 winlds[16 * 32 * 16];
  if (is_h) {
#pragma unroll
    for (int it = 0; it < I_ / 32; ++it) {
      v16bf b = load_b_t(WinT, I_, n0, it * 32, lane);
      *(v16bf*)&winlds[(it * 32 + lane) * 16] = b;
    }
  }
  __syncthreads();

  // Register-resident LIF state, in WMMA C-fragment layout:
  // slot = tile*8 + r  <->  element (m = tile*16 + mbase + r, n)
  float memv[16];
  float refv[16];
#pragma unroll
  for (int i = 0; i < 16; ++i) { memv[i] = 0.0f; refv[i] = 0.0f; }

  const float* xrow0 = x + (size_t)(lane & 15) * (S_ * I_);
  const float* xrow1 = xrow0 + (size_t)16 * (S_ * I_);
  const __bf16* srow0 = spikes + (size_t)(lane & 15) * H_;
  const __bf16* srow1 = srow0 + (size_t)16 * H_;
  const __bf16* Bmat = is_h ? LT : WhT;

  unsigned bar = 0;
  for (int t = 0; t < S_; ++t) {
    const float* xr0 = xrow0 + (size_t)t * I_;
    const float* xr1 = xrow1 + (size_t)t * I_;

    // ---------------- Phase A: ic = x_t @ W_in, LIF update, emit spikes ----
    if (is_h) {
      v8f c0 = {};
      v8f c1 = {};
#pragma unroll 4
      for (int it = 0; it < I_ / 32; ++it) {
        v16bf a0 = load_a_from_f32(xr0, it * 32, lane);
        v16bf a1 = load_a_from_f32(xr1, it * 32, lane);
        v16bf b  = *(const v16bf*)&winlds[(it * 32 + lane) * 16];
        c0 = __builtin_amdgcn_wmma_f32_16x16x32_bf16(false, a0, false, b,
                                                     (short)0, c0, false, false);
        c1 = __builtin_amdgcn_wmma_f32_16x16x32_bf16(false, a1, false, b,
                                                     (short)0, c1, false, false);
      }
      // Prefetch next timestep's activations while we do state math + GEMM B.
      __builtin_prefetch(xr0 + I_, 0, 1);
      __builtin_prefetch(xr1 + I_, 0, 1);
#pragma unroll
      for (int tile = 0; tile < 2; ++tile) {
#pragma unroll
        for (int r = 0; r < 8; ++r) {
          int slot = tile * 8 + r;
          int m = tile * 16 + mbase + r;
          float ic = tile ? c1[r] : c0[r];
          float rf = refv[slot];
          float mm = memv[slot];
          float active = (rf <= 0.0f) ? 1.0f : 0.0f;
          mm = DECAY_ * mm + ic * active;
          float sp = (mm >= THRESH_) ? 1.0f : 0.0f;
          mm = mm * (1.0f - sp);
          rf = fmaxf(rf - 1.0f, 0.0f) + sp * REFRAC_;
          memv[slot] = mm;
          refv[slot] = rf;
          spikes[(size_t)m * H_ + n] = (__bf16)sp;   // 0/1 exact in bf16
        }
      }
    }
    grid_barrier(cnt, gen, NBLK, ++bar);

    // ---------------- Phase B: spikes @ {L | W_h} --------------------------
    {
      v8f c0 = {};
      v8f c1 = {};
#pragma unroll 4
      for (int kk = 0; kk < H_; kk += 32) {
        v16bf a0 = load_a_bf16(srow0, kk, lane);
        v16bf a1 = load_a_bf16(srow1, kk, lane);
        v16bf b  = load_b_t(Bmat, H_, n0, kk, lane);
        c0 = __builtin_amdgcn_wmma_f32_16x16x32_bf16(false, a0, false, b,
                                                     (short)0, c0, false, false);
        c1 = __builtin_amdgcn_wmma_f32_16x16x32_bf16(false, a1, false, b,
                                                     (short)0, c1, false, false);
      }
      if (is_h) {
#pragma unroll
        for (int tile = 0; tile < 2; ++tile)
#pragma unroll
          for (int r = 0; r < 8; ++r)
            memv[tile * 8 + r] -= (tile ? c1[r] : c0[r]) * LATSCL_;
      } else {
#pragma unroll
        for (int tile = 0; tile < 2; ++tile) {
#pragma unroll
          for (int r = 0; r < 8; ++r) {
            int m = tile * 16 + mbase + r;
            out[(size_t)m * (S_ * O_) + (size_t)t * O_ + n] =
                tile ? c1[r] : c0[r];
          }
        }
      }
    }
    grid_barrier(cnt, gen, NBLK, ++bar);
  }
}

// ---------------------------------------------------------------------------
// Host launcher
// ---------------------------------------------------------------------------
extern "C" void kernel_launch(void* const* d_in, const int* in_sizes, int n_in,
                              void* d_out, int out_size, void* d_ws, size_t ws_size,
                              hipStream_t stream) {
  (void)in_sizes; (void)n_in; (void)out_size; (void)ws_size;

  const float* x   = (const float*)d_in[0];   // [B,S,I]
  const float* Win = (const float*)d_in[1];   // [I,H]
  const float* Wh  = (const float*)d_in[2];   // [H,O]
  const float* L   = (const float*)d_in[3];   // [H,H]
  float* out = (float*)d_out;                 // [B,S,O]

  // Workspace layout (~12.2 MB)
  char* ws = (char*)d_ws;
  __bf16*   WinT = (__bf16*)(ws + 0);            // [H][I]  2 MB
  __bf16*   LT   = (__bf16*)(ws + 2097152);      // [H][H]  8 MB
  __bf16*   WhT  = (__bf16*)(ws + 10485760);     // [O][H]  2 MB
  __bf16*   spk  = (__bf16*)(ws + 12582912);     // [B][H]  128 KB
  unsigned* sync = (unsigned*)(ws + 12713984);   // {cnt, gen}

  convert_transpose_kernel<<<(I_ * H_ + 255) / 256, 256, 0, stream>>>(Win, WinT, I_, H_);
  convert_transpose_kernel<<<(H_ * H_ + 255) / 256, 256, 0, stream>>>(L, LT, H_, H_);
  convert_transpose_kernel<<<(H_ * O_ + 255) / 256, 256, 0, stream>>>(Wh, WhT, H_, O_);
  zero_u32_kernel<<<1, 64, 0, stream>>>(sync, 2);

  // Single persistent kernel carries the whole 1024-step recurrence.
  snn_persistent<<<NBLK, 32, 0, stream>>>(x, WinT, LT, WhT, spk, out,
                                          sync + 0, sync + 1);
}
